// GCN_1511828488357
// MI455X (gfx1250) — matrix operand order repaired and checked by reading, compiled, not run on hardware
//
#include <hip/hip_runtime.h>
#include <hip/hip_bf16.h>
#include <cstdint>

typedef __attribute__((ext_vector_type(16))) _Float16 v16h;
typedef __attribute__((ext_vector_type(8)))  _Float16 v8h;
typedef __attribute__((ext_vector_type(8)))  float    v8f;

#define DFEAT 512
#define HID   16
#define NCLS  40
#define KSTEPS (DFEAT / 32)   // 16
#define BSTRIDE 24            // halves per lane-fragment: 48B, 16B-aligned, 2-way banks

// ---------------- init helpers ----------------
__global__ void fill_f32(float* p, float v, long long n) {
  long long i = (long long)blockIdx.x * blockDim.x + threadIdx.x;
  long long stride = (long long)gridDim.x * blockDim.x;
  for (; i < n; i += stride) p[i] = v;
}

// deg[dst] += 1 per edge (deg pre-filled with 1.0 for self-loop)
__global__ void degree_k(const int* __restrict__ dst, float* __restrict__ deg, int E) {
  int e = blockIdx.x * blockDim.x + threadIdx.x;
  if (e < E) atomicAdd(&deg[dst[e]], 1.0f);
}

__global__ void dinv_k(float* deg, int n) {
  int i = blockIdx.x * blockDim.x + threadIdx.x;
  if (i < n) { float d = deg[i]; deg[i] = d > 0.f ? rsqrtf(d) : 0.f; }
}

// ---------------- layer-1 transform: h1 = x @ W1 (WMMA f16->f32) ----------------
__global__ void __launch_bounds__(256) gemm1_wmma(
    const float* __restrict__ x, const float* __restrict__ W,
    float* __restrict__ h1, int n) {
  // W1 pre-shuffled into per-(kstep,lane) B fragments: 16*32*24 halves = 24 KB
  __shared__ alignas(16) _Float16 sB[KSTEPS * 32 * BSTRIDE];

  int tid = threadIdx.x;
  // Stage: logical j = (kstep, lane, elem). B fragment layout (16-bit B 32x16):
  //   lanes 0-15 : elems 0-7 = K kk+0..7,  elems 8-15 = K kk+16..23
  //   lanes 16-31: elems 0-7 = K kk+8..15, elems 8-15 = K kk+24..31,  col = lane & 15
  for (int j = tid; j < KSTEPS * 32 * 16; j += 256) {
    int kstep = j >> 9;
    int rem   = j & 511;
    int lane  = rem >> 4;
    int elem  = rem & 15;
    int k   = kstep * 32 + ((lane >> 4) << 3) + (elem & 7) + ((elem >> 3) << 4);
    int col = lane & 15;
    sB[(kstep * 32 + lane) * BSTRIDE + elem] = (_Float16)W[k * HID + col];
  }
  __syncthreads();

  int wave = tid >> 5;
  int lane = tid & 31;
  long long tile = (long long)blockIdx.x * 8 + wave;   // 16-row tile of nodes
  long long row0 = tile * 16;
  if (row0 >= n) return;

  int hi   = lane >> 4;     // 0: lanes 0-15, 1: lanes 16-31
  int rsub = lane & 15;     // A-row within tile / D-column
  long long arow = row0 + rsub;
  if (arow >= n) arow = n - 1;                 // clamp (OOB rows never stored)
  const float* xr = x + arow * DFEAT;

  v8f acc = {};
  for (int kk = 0; kk < DFEAT; kk += 32) {
    // A fragment, 16-bit A 16x32 layout:
    //   lanes 0-15 : elems 0-7 = K kk+0..7,  elems 8-15 = K kk+16..23
    //   lanes 16-31: elems 0-7 = K kk+8..15, elems 8-15 = K kk+24..31
    int kb = kk + hi * 8;
    float4 f0 = *(const float4*)(xr + kb);
    float4 f1 = *(const float4*)(xr + kb + 4);
    float4 f2 = *(const float4*)(xr + kb + 16);
    float4 f3 = *(const float4*)(xr + kb + 20);
    v16h a;
    a[0]=(_Float16)f0.x;  a[1]=(_Float16)f0.y;  a[2]=(_Float16)f0.z;  a[3]=(_Float16)f0.w;
    a[4]=(_Float16)f1.x;  a[5]=(_Float16)f1.y;  a[6]=(_Float16)f1.z;  a[7]=(_Float16)f1.w;
    a[8]=(_Float16)f2.x;  a[9]=(_Float16)f2.y;  a[10]=(_Float16)f2.z; a[11]=(_Float16)f2.w;
    a[12]=(_Float16)f3.x; a[13]=(_Float16)f3.y; a[14]=(_Float16)f3.z; a[15]=(_Float16)f3.w;

    // B fragment: two ds_load_b128 from the pre-shuffled LDS image
    const _Float16* bp = &sB[(((kk >> 5) << 5) + lane) * BSTRIDE];
    v8h bl = *(const v8h*)(bp);
    v8h bh = *(const v8h*)(bp + 8);
    v16h b = __builtin_shufflevector(bl, bh, 0,1,2,3,4,5,6,7,8,9,10,11,12,13,14,15);

    __builtin_prefetch(xr + kk + 32, 0, 1);  // speculative global_prefetch_b8

    acc = __builtin_amdgcn_wmma_f32_16x16x32_f16(
        /*neg_a=*/false, a, /*neg_b=*/false, b,
        /*c_mod=*/(short)0, acc, /*reuse_a=*/false, /*reuse_b=*/false);
  }

  // D layout: VGPR r -> row (r + 8*hi), col rsub
  int mbase = hi * 8;
  #pragma unroll
  for (int r = 0; r < 8; ++r) {
    long long m = row0 + mbase + r;
    if (m < n) h1[m * HID + rsub] = acc[r];
  }
}

// ---------------- edge scatter layer 1 (L2-resident atomics) ----------------
__global__ void scatter1_k(const int* __restrict__ src, const int* __restrict__ dst,
                           const float* __restrict__ dinv, const float* __restrict__ h1,
                           float* __restrict__ agg, int E) {
  int e = blockIdx.x * blockDim.x + threadIdx.x;
  if (e >= E) return;
  int s = src[e], d = dst[e];
  float nrm = dinv[s] * dinv[d];
  const float4* hs = (const float4*)(h1 + (long long)s * HID);
  float* ag = agg + (long long)d * HID;
  #pragma unroll
  for (int j = 0; j < 4; ++j) {
    float4 v = hs[j];
    atomicAdd(ag + 4 * j + 0, v.x * nrm);
    atomicAdd(ag + 4 * j + 1, v.y * nrm);
    atomicAdd(ag + 4 * j + 2, v.z * nrm);
    atomicAdd(ag + 4 * j + 3, v.w * nrm);
  }
}

// self-loop + bias + relu, in place on agg1
__global__ void node1_k(const float* __restrict__ h1, const float* __restrict__ dinv,
                        const float* __restrict__ b1, float* agg, long long n16) {
  long long i = (long long)blockIdx.x * blockDim.x + threadIdx.x;
  if (i >= n16) return;
  long long node = i >> 4;
  int f = (int)(i & 15);
  float dd = dinv[node];
  float v = agg[i] + h1[i] * dd * dd + b1[f];
  agg[i] = v > 0.f ? v : 0.f;
}

// ---------------- layer-2 transform: h2 = hrelu @ W2 (tiny, scalar) ----------------
__global__ void gemm2_k(const float* __restrict__ hr, const float* __restrict__ W2,
                        float* __restrict__ h2, long long ncls_total) {
  long long i = (long long)blockIdx.x * blockDim.x + threadIdx.x;
  if (i >= ncls_total) return;
  long long node = i / NCLS;
  int c = (int)(i % NCLS);
  const float* row = hr + node * HID;
  float acc = 0.f;
  #pragma unroll
  for (int k = 0; k < HID; ++k) acc += row[k] * W2[k * NCLS + c];
  h2[i] = acc;
}

__global__ void scatter2_k(const int* __restrict__ src, const int* __restrict__ dst,
                           const float* __restrict__ dinv, const float* __restrict__ h2,
                           float* __restrict__ out, int E) {
  int e = blockIdx.x * blockDim.x + threadIdx.x;
  if (e >= E) return;
  int s = src[e], d = dst[e];
  float nrm = dinv[s] * dinv[d];
  const float* hs = h2 + (long long)s * NCLS;
  float* od = out + (long long)d * NCLS;
  #pragma unroll 8
  for (int c = 0; c < NCLS; ++c) atomicAdd(od + c, hs[c] * nrm);
}

// self-loop + bias + log_softmax, in place on out
__global__ void node2_k(const float* __restrict__ h2, const float* __restrict__ dinv,
                        const float* __restrict__ b2, float* out, int n) {
  int node = blockIdx.x * blockDim.x + threadIdx.x;
  if (node >= n) return;
  float dd = dinv[node];
  float sn = dd * dd;
  const float* hs = h2 + (long long)node * NCLS;
  float* o = out + (long long)node * NCLS;
  float row[NCLS];
  float mx = -1e30f;
  #pragma unroll
  for (int c = 0; c < NCLS; ++c) {
    row[c] = o[c] + hs[c] * sn + b2[c];
    mx = fmaxf(mx, row[c]);
  }
  float sum = 0.f;
  #pragma unroll
  for (int c = 0; c < NCLS; ++c) sum += __expf(row[c] - mx);
  float lg = __logf(sum);
  #pragma unroll
  for (int c = 0; c < NCLS; ++c) o[c] = row[c] - mx - lg;
}

extern "C" void kernel_launch(void* const* d_in, const int* in_sizes, int n_in,
                              void* d_out, int out_size, void* d_ws, size_t ws_size,
                              hipStream_t stream) {
  (void)n_in; (void)out_size; (void)ws_size;
  const float* x  = (const float*)d_in[0];
  const int*   ei = (const int*)d_in[1];
  const float* W1 = (const float*)d_in[2];
  const float* b1 = (const float*)d_in[3];
  const float* W2 = (const float*)d_in[4];
  const float* b2 = (const float*)d_in[5];
  float* out = (float*)d_out;

  int n = in_sizes[0] / DFEAT;     // 100000
  int E = in_sizes[1] / 2;         // 3200000
  const int* src = ei;
  const int* dst = ei + E;

  float* ws   = (float*)d_ws;
  float* dinv = ws;                                   // n
  float* h1   = ws + n;                               // 16n
  float* agg1 = h1 + (long long)n * HID;              // 16n (becomes hrelu)
  float* h2   = agg1 + (long long)n * HID;            // 40n

  const int T = 256;
  long long n16 = (long long)n * HID;
  long long nc  = (long long)n * NCLS;

  // init accumulators (d_ws/d_out are poisoned; must zero each call)
  fill_f32<<<256,  T, 0, stream>>>(dinv, 1.0f, n);    // self-loop degree
  fill_f32<<<512,  T, 0, stream>>>(agg1, 0.0f, n16);
  fill_f32<<<1024, T, 0, stream>>>(out,  0.0f, nc);

  degree_k<<<(E + T - 1) / T, T, 0, stream>>>(dst, dinv, E);
  dinv_k  <<<(n + T - 1) / T, T, 0, stream>>>(dinv, n);

  int tiles = (n + 15) / 16;
  gemm1_wmma<<<(tiles + 7) / 8, 256, 0, stream>>>(x, W1, h1, n);

  scatter1_k<<<(E + T - 1) / T, T, 0, stream>>>(src, dst, dinv, h1, agg1, E);
  node1_k   <<<(int)((n16 + T - 1) / T), T, 0, stream>>>(h1, dinv, b1, agg1, n16);

  gemm2_k   <<<(int)((nc + T - 1) / T), T, 0, stream>>>(agg1, W2, h2, nc);
  scatter2_k<<<(E + T - 1) / T, T, 0, stream>>>(src, dst, dinv, h2, out, E);
  node2_k   <<<(n + T - 1) / T, T, 0, stream>>>(h2, dinv, b2, out, n);
}
